// MambaModel_16320875724853
// MI455X (gfx1250) — compile-verified, matrix operand order checked by reading, not run on hardware
//
#include <hip/hip_runtime.h>
#include <math.h>

#define L_SEQ   2048
#define D_MODEL 1024
#define D_INNER 2048
#define D_STATE 16
#define DT_RANK 64
#define N_LAYER 4

#define BK   512            // K-chunk staged in LDS
#define BKP  (BK + 8)       // padded LDS row stride (bank-conflict avoidance)

typedef __attribute__((ext_vector_type(16))) __bf16 v16bf;
typedef __attribute__((ext_vector_type(8)))  __bf16 v8bf;
typedef __attribute__((ext_vector_type(8)))  float  v8f;

// ---------------------------------------------------------------------------
// Fragment loaders.  ISA 7.12.2 (wave32):
//  A (16x32 bf16):  lanes 0-15 -> M=0..15 hold K = {0..7, 16..23};
//                   lanes 16-31 -> same M hold K = {8..15, 24..31}.
//  B (32x16 bf16):  mirrored with lane = N.
//  C/D (16x16 f32): lane = N; VGPR j -> M = j (lanes 0-15) / M = 8+j (16-31).
// ---------------------------------------------------------------------------

__device__ inline v16bf frag_from_bf16(const __bf16* base, int row, int ld,
                                       int k, int hf) {
  const __bf16* p = base + (size_t)row * ld + k + hf * 8;
  v8bf a0 = *(const v8bf*)p;
  v8bf a1 = *(const v8bf*)(p + 16);
  v16bf v;
#pragma unroll
  for (int i = 0; i < 8; ++i) { v[i] = a0[i]; v[8 + i] = a1[i]; }
  return v;
}

__device__ inline v16bf frag_from_f32(const float* base, int row, int ld,
                                      int k, int hf) {
  const float* p = base + (size_t)row * ld + k + hf * 8;
  v16bf v;
#pragma unroll
  for (int i = 0; i < 8; ++i) {
    v[i]     = (__bf16)p[i];
    v[8 + i] = (__bf16)p[16 + i];
  }
  return v;
}

__device__ inline v16bf frag_from_lds(const __bf16* Bs, int r, int k, int hf) {
  const __bf16* p = Bs + r * BKP + k + hf * 8;
  v8bf b0 = *(const v8bf*)p;          // ds_load_b128
  v8bf b1 = *(const v8bf*)(p + 16);
  v16bf v;
#pragma unroll
  for (int i = 0; i < 8; ++i) { v[i] = b0[i]; v[8 + i] = b1[i]; }
  return v;
}

// ---------------------------------------------------------------------------
// Generic NT GEMM:  C[M x N] = A[M x K] * B[N x K]^T   (fp32 accumulate)
// 256 threads = 8 waves; wave computes 64x16 (4 M-tiles, B frag reused).
// B tile (16 x BK bf16) staged in LDS, shared by all 8 waves.
// EPI: 0 = none, 1 = softplus(x + bias[n])
// ---------------------------------------------------------------------------
template <bool A_IS_BF16, int EPI>
__global__ __launch_bounds__(256)
void gemm_wmma(const void* __restrict__ Aptr, const __bf16* __restrict__ Bw,
               float* __restrict__ C, const float* __restrict__ bias,
               int K, int lda, int ldb, int ldc) {
  __shared__ __bf16 Bs[16 * BKP];

  const int lane  = threadIdx.x & 31;
  const int wave  = threadIdx.x >> 5;
  const int r     = lane & 15;
  const int hf    = lane >> 4;
  const int mWave = blockIdx.y * 512 + wave * 64;
  const int nBase = blockIdx.x * 16;

  v8f acc[4] = {};

  for (int k0 = 0; k0 < K; k0 += BK) {
    const int kc = (K - k0 < BK) ? (K - k0) : BK;   // multiple of 32 here
    __syncthreads();
    // cooperative stage of B tile: 16 rows x kc bf16, 8-elt (16B) chunks
    const int nchunk = 16 * (kc >> 3);
    for (int idx = threadIdx.x; idx < nchunk; idx += 256) {
      const int row = idx / (kc >> 3);
      const int c   = idx - row * (kc >> 3);
      *(v8bf*)&Bs[row * BKP + c * 8] =
          *(const v8bf*)&Bw[(size_t)(nBase + row) * ldb + k0 + c * 8];
    }
    // prefetch next B chunk into cache while we compute (global_prefetch_b8)
    if (k0 + BK < K)
      __builtin_prefetch(&Bw[(size_t)(nBase + r) * ldb + k0 + BK], 0, 1);
    __syncthreads();

    for (int k = 0; k < kc; k += 32) {
      v16bf b = frag_from_lds(Bs, r, k, hf);
#pragma unroll
      for (int t = 0; t < 4; ++t) {
        v16bf a;
        if (A_IS_BF16)
          a = frag_from_bf16((const __bf16*)Aptr, mWave + t * 16 + r, lda,
                             k0 + k, hf);
        else
          a = frag_from_f32((const float*)Aptr, mWave + t * 16 + r, lda,
                            k0 + k, hf);
        acc[t] = __builtin_amdgcn_wmma_f32_16x16x32_bf16(
            false, a, false, b, (short)0, acc[t], false, false);
      }
    }
  }

  const int n = nBase + r;
#pragma unroll
  for (int t = 0; t < 4; ++t) {
#pragma unroll
    for (int j = 0; j < 8; ++j) {
      int m   = mWave + t * 16 + hf * 8 + j;
      float v = acc[t][j];
      if (EPI == 1) {
        v += bias[n];
        v = (v > 20.f) ? v : logf(1.f + expf(v));   // softplus
      }
      C[(size_t)m * ldc + n] = v;
    }
  }
}

// ---------------------------------------------------------------------------
// fp32 -> bf16 weight conversion (weights stay L2-resident; one cheap pass)
// ---------------------------------------------------------------------------
__global__ __launch_bounds__(256)
void cvt_bf16_kernel(const float* __restrict__ in, __bf16* __restrict__ out,
                     int n) {
  const int i = blockIdx.x * 256 + threadIdx.x;
  if (i < n) out[i] = (__bf16)in[i];
}

// ---------------------------------------------------------------------------
// residual add + RMSNorm, emits bf16 normalized activations.  1 block / row.
// ---------------------------------------------------------------------------
__global__ __launch_bounds__(256)
void resnorm_kernel(const float* __restrict__ hid, float* __restrict__ res,
                    __bf16* __restrict__ hout, const float* __restrict__ nw,
                    int first) {
  const int row = blockIdx.x;
  const float* hp = hid + (size_t)row * D_MODEL;
  float*       rp = res + (size_t)row * D_MODEL;
  __bf16*      op = hout + (size_t)row * D_MODEL;

  float vals[4];
  float ss = 0.f;
#pragma unroll
  for (int i = 0; i < 4; ++i) {
    int e   = threadIdx.x + i * 256;
    float v = hp[e] + (first ? 0.f : rp[e]);
    rp[e]   = v;
    vals[i] = v;
    ss += v * v;
  }
#pragma unroll
  for (int off = 16; off >= 1; off >>= 1) ss += __shfl_xor(ss, off);
  __shared__ float smem[8];
  if ((threadIdx.x & 31) == 0) smem[threadIdx.x >> 5] = ss;
  __syncthreads();
  float tot = 0.f;
#pragma unroll
  for (int w = 0; w < 8; ++w) tot += smem[w];
  const float scale = rsqrtf(tot / (float)D_MODEL + 1e-5f);
#pragma unroll
  for (int i = 0; i < 4; ++i) {
    int e = threadIdx.x + i * 256;
    op[e] = (__bf16)(vals[i] * scale * nw[e]);
  }
}

// ---------------------------------------------------------------------------
// causal depthwise conv1d (K=4) + SiLU.  u = xz[:, :D_INNER].
// ---------------------------------------------------------------------------
__global__ __launch_bounds__(256)
void conv_silu_kernel(const float* __restrict__ xz, const float* __restrict__ cw,
                      const float* __restrict__ cb, float* __restrict__ ucf,
                      __bf16* __restrict__ ucb) {
  const int idx = blockIdx.x * 256 + threadIdx.x;   // over L*D_INNER
  const int d   = idx & (D_INNER - 1);
  const int l   = idx >> 11;
  float acc = cb[d];
#pragma unroll
  for (int j = 0; j < 4; ++j) {
    int ll = l - 3 + j;
    if (ll >= 0) acc += xz[(size_t)ll * (2 * D_INNER) + d] * cw[d * 4 + j];
  }
  float s = acc / (1.f + expf(-acc));               // silu
  ucf[idx] = s;
  ucb[idx] = (__bf16)s;
}

// ---------------------------------------------------------------------------
// selective scan + D skip + SiLU(z) gate.  Wave32: lanes 0-15 = channel d0,
// lanes 16-31 = channel d0+1; lane&15 = state index n.  16-lane butterfly
// reduces h*C over d_state.  Sequential over L.
// ---------------------------------------------------------------------------
__global__ __launch_bounds__(256)
void scan_kernel(const float* __restrict__ delta, const float* __restrict__ ucf,
                 const float* __restrict__ xdbl, const float* __restrict__ xz,
                 const float* __restrict__ Alog, const float* __restrict__ Dp,
                 __bf16* __restrict__ ybf) {
  const int lane = threadIdx.x & 31;
  const int wave = threadIdx.x >> 5;
  const int n    = lane & 15;
  const int d    = blockIdx.x * 16 + wave * 2 + (lane >> 4);

  const float A  = -expf(Alog[d * D_STATE + n]);
  const float Dd = Dp[d];
  float h = 0.f;

  for (int l = 0; l < L_SEQ; ++l) {
    const float dt = delta[(size_t)l * D_INNER + d];
    const float u  = ucf[(size_t)l * D_INNER + d];
    const float Bn = xdbl[(size_t)l * 96 + DT_RANK + n];
    const float Cn = xdbl[(size_t)l * 96 + DT_RANK + D_STATE + n];
    h = expf(dt * A) * h + dt * u * Bn;
    float yp = h * Cn;
#pragma unroll
    for (int off = 1; off < 16; off <<= 1) yp += __shfl_xor(yp, off);
    if (n == 0) {
      const float z = xz[(size_t)l * (2 * D_INNER) + D_INNER + d];
      const float y = (yp + u * Dd) * (z / (1.f + expf(-z)));
      ybf[(size_t)l * D_INNER + d] = (__bf16)y;
    }
  }
}

__global__ __launch_bounds__(256)
void final_add_kernel(const float* __restrict__ hid, const float* __restrict__ res,
                      float* __restrict__ out) {
  const int i = blockIdx.x * 256 + threadIdx.x;
  out[i] = hid[i] + res[i];
}

// ---------------------------------------------------------------------------
extern "C" void kernel_launch(void* const* d_in, const int* in_sizes, int n_in,
                              void* d_out, int out_size, void* d_ws, size_t ws_size,
                              hipStream_t stream) {
  const float* hidden_in = (const float*)d_in[0];
  const float* norm_w    = (const float*)d_in[1];
  const float* ipw       = (const float*)d_in[2];   // (4, 4096, 1024)
  const float* cw        = (const float*)d_in[3];   // (4, 2048, 4)
  const float* cb        = (const float*)d_in[4];   // (4, 2048)
  const float* xpw       = (const float*)d_in[5];   // (4, 96, 2048)
  const float* dtw       = (const float*)d_in[6];   // (4, 2048, 64)
  const float* dtb       = (const float*)d_in[7];   // (4, 2048)
  const float* Alog      = (const float*)d_in[8];   // (4, 2048, 16)
  const float* Dp        = (const float*)d_in[9];   // (4, 2048)
  const float* opw       = (const float*)d_in[10];  // (4, 1024, 2048)

  char* ws = (char*)d_ws;
  float*  res       = (float*)ws;  ws += (size_t)L_SEQ * D_MODEL * 4;
  float*  hiddenBuf = (float*)ws;  ws += (size_t)L_SEQ * D_MODEL * 4;
  __bf16* hbf       = (__bf16*)ws; ws += (size_t)L_SEQ * D_MODEL * 2;
  float*  xz        = (float*)ws;  ws += (size_t)L_SEQ * 2 * D_INNER * 4;
  float*  ucf       = (float*)ws;  ws += (size_t)L_SEQ * D_INNER * 4;
  __bf16* ucb       = (__bf16*)ws; ws += (size_t)L_SEQ * D_INNER * 2;
  float*  xdbl      = (float*)ws;  ws += (size_t)L_SEQ * 96 * 4;
  float*  delta     = (float*)ws;  ws += (size_t)L_SEQ * D_INNER * 4;
  __bf16* ybf       = (__bf16*)ws; ws += (size_t)L_SEQ * D_INNER * 2;

  const int n_ipw = N_LAYER * 2 * D_INNER * D_MODEL;   // 16.8M
  const int n_xpw = N_LAYER * 96 * D_INNER;            // 0.79M
  const int n_dtw = N_LAYER * D_INNER * DT_RANK;       // 0.52M
  const int n_opw = N_LAYER * D_MODEL * D_INNER;       // 8.4M
  __bf16* ipw_bf = (__bf16*)ws; ws += (size_t)n_ipw * 2;
  __bf16* xpw_bf = (__bf16*)ws; ws += (size_t)n_xpw * 2;
  __bf16* dtw_bf = (__bf16*)ws; ws += (size_t)n_dtw * 2;
  __bf16* opw_bf = (__bf16*)ws; ws += (size_t)n_opw * 2;

  const dim3 blk(256);

  // one-time (per call) bf16 weight staging
  cvt_bf16_kernel<<<(n_ipw + 255) / 256, blk, 0, stream>>>(ipw, ipw_bf, n_ipw);
  cvt_bf16_kernel<<<(n_xpw + 255) / 256, blk, 0, stream>>>(xpw, xpw_bf, n_xpw);
  cvt_bf16_kernel<<<(n_dtw + 255) / 256, blk, 0, stream>>>(dtw, dtw_bf, n_dtw);
  cvt_bf16_kernel<<<(n_opw + 255) / 256, blk, 0, stream>>>(opw, opw_bf, n_opw);

  for (int i = 0; i < N_LAYER; ++i) {
    const float* hsrc = (i == 0) ? hidden_in : hiddenBuf;

    resnorm_kernel<<<L_SEQ, blk, 0, stream>>>(hsrc, res, hbf,
                                              norm_w + i * D_MODEL, i == 0);

    // in_proj: (2048 x 4096) = h(2048x1024) . W^T
    gemm_wmma<true, 0><<<dim3(2 * D_INNER / 16, L_SEQ / 512), blk, 0, stream>>>(
        hbf, ipw_bf + (size_t)i * 2 * D_INNER * D_MODEL, xz, nullptr,
        D_MODEL, D_MODEL, D_MODEL, 2 * D_INNER);

    conv_silu_kernel<<<(L_SEQ * D_INNER) / 256, blk, 0, stream>>>(
        xz, cw + (size_t)i * D_INNER * 4, cb + (size_t)i * D_INNER, ucf, ucb);

    // x_proj: (2048 x 96) = uc(2048x2048) . W^T
    gemm_wmma<true, 0><<<dim3(96 / 16, L_SEQ / 512), blk, 0, stream>>>(
        ucb, xpw_bf + (size_t)i * 96 * D_INNER, xdbl, nullptr,
        D_INNER, D_INNER, D_INNER, 96);

    // dt_proj + softplus: (2048 x 2048) = dt(2048x64) . W^T + b
    gemm_wmma<false, 1><<<dim3(D_INNER / 16, L_SEQ / 512), blk, 0, stream>>>(
        xdbl, dtw_bf + (size_t)i * D_INNER * DT_RANK, delta,
        dtb + (size_t)i * D_INNER, DT_RANK, 96, DT_RANK, D_INNER);

    scan_kernel<<<D_INNER / 16, blk, 0, stream>>>(
        delta, ucf, xdbl, xz, Alog + (size_t)i * D_INNER * D_STATE,
        Dp + (size_t)i * D_INNER, ybf);

    // out_proj: (2048 x 1024) = y(2048x2048) . W^T
    gemm_wmma<true, 0><<<dim3(D_MODEL / 16, L_SEQ / 512), blk, 0, stream>>>(
        ybf, opw_bf + (size_t)i * D_MODEL * D_INNER, hiddenBuf, nullptr,
        D_INNER, D_INNER, D_INNER, D_MODEL);
  }

  final_add_kernel<<<(L_SEQ * D_MODEL) / 256, blk, 0, stream>>>(
      hiddenBuf, res, (float*)d_out);
}
